// SimpleRNN_46514495815958
// MI455X (gfx1250) — matrix operand order repaired and checked by reading, not compiled, not run on hardware
//
#include <hip/hip_runtime.h>
#include <math.h>

#define TSTEPS    28
#define INPUT_DIM 28
#define KPAD      32
#define HID       128
#define CLASSES   10
#define BTILE     32
#define THREADS   256

// Padded LDS row strides (elements). Hot-path strides mod 64 dwords == 4 -> a
// half-wave's 16 b128 accesses (4 dwords each, rows m=0..15) tile banks 0..63 once.
#define XLD  (TSTEPS * KPAD + 8)   // 904 elem = 1808 B per batch row (hot)
#define HLD  (HID + 8)             // 136 elem = 272 B per h row       (hot)
#define WLD  (HID + 4)             // 132 elem = 264 B per W_hh row    (init-only)

typedef __attribute__((ext_vector_type(16))) __bf16 v16bf;
typedef __attribute__((ext_vector_type(8)))  __bf16 v8bf;
typedef __attribute__((ext_vector_type(4)))  __bf16 v4bf;
typedef __attribute__((ext_vector_type(8)))  float  v8f;
typedef __attribute__((ext_vector_type(4)))  float  v4f;

// ---- dynamic LDS layout (bytes), all 16B aligned ----
// Total 109056 B/WG -> 3 workgroups per WGP (327168 <= 327680), 6 waves/SIMD.
constexpr int SMEM_XS_OFF = 0;
constexpr int SMEM_XS_SZ  = BTILE * XLD * 2;                   // 57856
constexpr int SMEM_WH_OFF = SMEM_XS_OFF + SMEM_XS_SZ;
constexpr int SMEM_WH_SZ  = HID * WLD * 2;                     // 33792
constexpr int SMEM_HF_OFF = SMEM_WH_OFF;                       // alias: f32 h_last [BTILE][HID]
constexpr int SMEM_H_OFF  = SMEM_WH_OFF + SMEM_WH_SZ;
constexpr int SMEM_H_SZ   = 2 * BTILE * HLD * 2;               // 17408
constexpr int SMEM_TOTAL  = SMEM_H_OFF + SMEM_H_SZ;            // 109056

static __device__ __forceinline__ __bf16 f2bf(float f) {
  return static_cast<__bf16>(f);   // native fptrunc; pairs fuse to v_cvt_pk_bf16_f32
}

static __device__ __forceinline__ v4bf f2bf4(v4f w) {
  v4bf p;
  p[0] = f2bf(w[0]); p[1] = f2bf(w[1]); p[2] = f2bf(w[2]); p[3] = f2bf(w[3]);
  return p;  // 2x v_cvt_pk_bf16_f32
}

static __device__ __forceinline__ float fast_tanh(float x) {
#if __has_builtin(__builtin_amdgcn_tanhf)
  return __builtin_amdgcn_tanhf(x);               // v_tanh_f32 (TRANS, co-executes)
#else
  float ax = __builtin_fabsf(x);
  float e  = __builtin_amdgcn_exp2f(ax * 2.8853900817779268f);
  float r  = 1.0f - 2.0f * __builtin_amdgcn_rcpf(e + 1.0f);
  return __builtin_copysignf(r, x);
#endif
}

// A-operand (16x32 bf16, row-major source). Lane L holds row m=L%16;
// elements 0..15 -> K = sel+0..15, sel=(L>=16)?16:0 (contiguous).
static __device__ __forceinline__ v16bf load_a_frag(const __bf16* buf, int m, int ld,
                                                    int kb, int lane) {
  const __bf16* p = buf + m * ld + kb + ((lane >= 16) ? 16 : 0);
  v8bf lo = *(const v8bf*)(p);        // ds_load_b128
  v8bf hi = *(const v8bf*)(p + 8);    // ds_load_b128
  v16bf a;
#pragma unroll
  for (int i = 0; i < 8; ++i) { a[i] = lo[i]; a[i + 8] = hi[i]; }
  return a;
}

// B-operand (32x16 bf16) from W^T view: B[k][n] = W[n][k], row-major W in LDS.
// Lanes 0-15 hold K=0..15, lanes 16-31 hold K=16..31 (contiguous).
static __device__ __forceinline__ v16bf load_b_frag(const __bf16* buf, int n, int ld,
                                                    int kb, int lane) {
  const __bf16* p = buf + n * ld + kb + ((lane >= 16) ? 16 : 0);
  v8bf lo = *(const v8bf*)(p);
  v8bf hi = *(const v8bf*)(p + 8);
  v16bf b;
#pragma unroll
  for (int i = 0; i < 8; ++i) { b[i] = lo[i]; b[i + 8] = hi[i]; }
  return b;
}

__global__ __launch_bounds__(THREADS)
__attribute__((amdgpu_waves_per_eu(6)))   // LDS caps us at 6 waves/SIMD anyway:
void rnn_fused_wmma_kernel(const float* __restrict__ x,   // give RA the full ~170 VGPRs
                           const float* __restrict__ W_ih,
                           const float* __restrict__ W_hh,
                           const float* __restrict__ b_ih,
                           const float* __restrict__ b_hh,
                           const float* __restrict__ W_fc,
                           const float* __restrict__ b_fc,
                           float* __restrict__ out) {
  extern __shared__ __align__(16) char smem[];
  __bf16* xs_lds = (__bf16*)(smem + SMEM_XS_OFF);   // [BTILE][XLD]  ([m][t*KPAD+k])
  __bf16* wh_lds = (__bf16*)(smem + SMEM_WH_OFF);   // [HID][WLD]
  float*  hf_lds = (float*)(smem + SMEM_HF_OFF);    // [BTILE][HID] (aliases wh_lds)
  __bf16* h_lds  = (__bf16*)(smem + SMEM_H_OFF);    // [2][BTILE][HLD]

  const int tid  = threadIdx.x;
  const int wave = tid >> 5;
  const int lane = tid & 31;
  const int b0   = blockIdx.x * BTILE;

  // ---- stage x tile: float4 global loads -> packed bf16 -> ds_store_b64 ----
  // Each timestep row is exactly 7 float4 (28 floats); 196 float4 per batch row.
  {
    const v4f* x4 = (const v4f*)(x + (size_t)b0 * (TSTEPS * INPUT_DIM));
    for (int i = tid; i < BTILE * 196; i += THREADS) {
      int m = i / 196;
      int r = i - m * 196;
      int t = r / 7;
      int k = (r - t * 7) * 4;
      *(v4bf*)(xs_lds + m * XLD + t * KPAD + k) = f2bf4(x4[i]);
    }
    v4bf z = {};
    for (int i = tid; i < BTILE * TSTEPS; i += THREADS) {   // zero K-pad 28..31
      int m = i / TSTEPS, t = i - m * TSTEPS;
      *(v4bf*)(xs_lds + m * XLD + t * KPAD + INPUT_DIM) = z;
    }
    // ---- stage W_hh as bf16 (padded rows), float4 loads ----
    const v4f* w4 = (const v4f*)W_hh;
    for (int i = tid; i < HID * (HID / 4); i += THREADS) {
      int nrow = i >> 5;
      int k = (i & 31) * 4;
      *(v4bf*)(wh_lds + nrow * WLD + k) = f2bf4(w4[i]);
    }
    // ---- h(0) = 0 ----
    for (int i = tid; i < (BTILE * HLD) / 4; i += THREADS)
      *(v4bf*)(h_lds + 4 * i) = z;
  }
  __syncthreads();

  // ---- per-wave resident weight fragments ----
  const int n = wave * 16 + (lane & 15);

  v16bf bwih;  // B[k][n] = W_ih[n][k], K padded to 32 with zeros
  {
    int kb = (lane >= 16) ? 16 : 0;
#pragma unroll
    for (int e = 0; e < 16; ++e) {
      int k = kb + e;
      bwih[e] = (k < INPUT_DIM) ? f2bf(W_ih[n * INPUT_DIM + k]) : f2bf(0.0f);
    }
  }
  v16bf bwh[4];  // B[k][n] = W_hh[n][k], 4 K-chunks of 32, resident in VGPRs
#pragma unroll
  for (int c = 0; c < 4; ++c) bwh[c] = load_b_frag(wh_lds, n, WLD, 32 * c, lane);

  const float biasv = b_ih[n] + b_hh[n];
  v8f biasc;
#pragma unroll
  for (int i = 0; i < 8; ++i) biasc[i] = biasv;

  const int m0   = (lane & 15);
  const int m1   = 16 + (lane & 15);
  const int mrow = (lane >= 16) ? 8 : 0;
  int cur = 0;
  v8f acc0, acc1;

  // One timestep's matmuls: fragment loads issued up front, then 10 WMMAs.
  auto step_mm = [&](int t) {
    const __bf16* hcur  = h_lds + cur * (BTILE * HLD);
    const __bf16* xbase = xs_lds + t * KPAD;
    v16bf ax0 = load_a_frag(xbase, m0, XLD, 0, lane);
    v16bf ax1 = load_a_frag(xbase, m1, XLD, 0, lane);
    v16bf ah0[4], ah1[4];
#pragma unroll
    for (int c = 0; c < 4; ++c) {
      ah0[c] = load_a_frag(hcur, m0, HLD, 32 * c, lane);
      ah1[c] = load_a_frag(hcur, m1, HLD, 32 * c, lane);
    }
    acc0 = __builtin_amdgcn_wmma_f32_16x16x32_bf16(false, ax0, false, bwih,
                                                   (short)0, biasc, false, false);
    acc1 = __builtin_amdgcn_wmma_f32_16x16x32_bf16(false, ax1, false, bwih,
                                                   (short)0, biasc, false, false);
#pragma unroll
    for (int c = 0; c < 4; ++c) {
      acc0 = __builtin_amdgcn_wmma_f32_16x16x32_bf16(false, ah0[c], false, bwh[c],
                                                     (short)0, acc0, false, false);
      acc1 = __builtin_amdgcn_wmma_f32_16x16x32_bf16(false, ah1[c], false, bwh[c],
                                                     (short)0, acc1, false, false);
    }
  };

  // ============ timesteps 0..26: tanh -> bf16 h (double-buffered) ============
  for (int t = 0; t < TSTEPS - 1; ++t) {
    step_mm(t);
    __bf16* hnxt = h_lds + (cur ^ 1) * (BTILE * HLD);
#pragma unroll
    for (int r = 0; r < 8; ++r) {
      float h0 = fast_tanh(acc0[r]);
      float h1 = fast_tanh(acc1[r]);
      hnxt[(mrow + r) * HLD + n]      = f2bf(h0);
      hnxt[(16 + mrow + r) * HLD + n] = f2bf(h1);
    }
    __syncthreads();
    cur ^= 1;
  }

  // ============ final timestep: keep f32 h_last for the FC head ============
  step_mm(TSTEPS - 1);
  __syncthreads();  // everyone done reading the aliased wh region and xs
#pragma unroll
  for (int r = 0; r < 8; ++r) {
    hf_lds[(mrow + r) * HID + n]      = fast_tanh(acc0[r]);
    hf_lds[(16 + mrow + r) * HID + n] = fast_tanh(acc1[r]);
  }
  __syncthreads();

  // ================= FC head: [BTILE,128] @ [10,128]^T + b =================
  for (int idx = tid; idx < BTILE * CLASSES; idx += THREADS) {
    int m = idx / CLASSES;
    int c = idx - m * CLASSES;
    const float* hrow = hf_lds + m * HID;
    const float* wrow = W_fc + c * HID;
    float s = b_fc[c];
#pragma unroll 8
    for (int k = 0; k < HID; ++k) s += hrow[k] * wrow[k];
    out[(size_t)(b0 + m) * CLASSES + c] = s;
  }
}

extern "C" void kernel_launch(void* const* d_in, const int* in_sizes, int n_in,
                              void* d_out, int out_size, void* d_ws, size_t ws_size,
                              hipStream_t stream) {
  (void)n_in; (void)out_size; (void)d_ws; (void)ws_size;
  const float* x    = (const float*)d_in[0];
  const float* W_ih = (const float*)d_in[1];
  const float* W_hh = (const float*)d_in[2];
  const float* b_ih = (const float*)d_in[3];
  const float* b_hh = (const float*)d_in[4];
  const float* W_fc = (const float*)d_in[5];
  const float* b_fc = (const float*)d_in[6];
  float* out = (float*)d_out;

  const int B    = in_sizes[0] / (TSTEPS * INPUT_DIM);  // 16384
  const int grid = B / BTILE;                           // 512 workgroups
  rnn_fused_wmma_kernel<<<grid, THREADS, SMEM_TOTAL, stream>>>(
      x, W_ih, W_hh, b_ih, b_hh, W_fc, b_fc, out);
}
